// VICTOR_v6_33474975105230
// MI455X (gfx1250) — matrix operand order repaired and checked by reading, not compiled, not run on hardware
//
#include <hip/hip_runtime.h>

// ---------------------------------------------------------------------------
// MI455X (gfx1250) implementation.
//
//   K0 pack_kernel  : fp32 weights -> fp16 pre-swizzled into WMMA B-fragment
//                     order (32 lanes x 16 contiguous halves per 32x16 tile),
//                     so every B operand is two 16-byte loads.
//   K1 feat_kernel  : hash-grid (16+8 levels) + SO(2) features -> 56 (pad 64) f16
//   K2 siren_gemm<2>: H1 = sin(om1*(X @ W1 + b1))   [v_wmma_f32_16x16x32_f16]
//   K3 siren_gemm<8>: Tr = sin(om2*(H1 @ W2 + b2))  [WMMA, A regs reused]
//   K4 heads_kernel : eps = mean_m softplus(gelu(Tr@mW1[m]+b)@mW2[m]+b)
//                     [WMMA + ds_add_f32 reduction]
//   K5 final_kernel : GNN refinement. LayerNorm over a size-1 feature axis is
//                     exactly (y - y)/sqrt(0+1e-6)*s + b = ln_bias[l], so
//                     out = softplus(ln_bias[1]) * sigmoid((1-psi)*50) exactly.
// Transcendentals on the (output-dead) heavy path use the HW trans unit
// (__sinf/__expf/__logf -> v_sin/v_exp/v_log, co-issuable with WMMA).
// ---------------------------------------------------------------------------

#define NN 65536
#define TC 8192
#define TF 4096

typedef __attribute__((ext_vector_type(16))) _Float16 v16h;
typedef __attribute__((ext_vector_type(8)))  _Float16 v8h;
typedef __attribute__((ext_vector_type(8)))  float    v8f;

union AV { v16h v; v8h h8[2]; _Float16 h[16]; };

__constant__ int RES_TAB[16] = {16,22,30,42,58,80,110,152,210,290,400,553,763,1053,1453,2005};

// ---- K0: fp32 -> fp16, pre-swizzled into WMMA fragment order ---------------
// Tile (ktile,otile) of a KxO weight: packed[t*512 + lane*16 + j] where
// t = tile index, k = ktile*32 + (lane&16 ? 8:0) + (j<8 ? j : j+8), o = otile*16 + (lane&15).
__global__ void pack_kernel(const float* __restrict__ W1, const float* __restrict__ W2,
                            const float* __restrict__ mW1,
                            _Float16* __restrict__ W1p, _Float16* __restrict__ W2p,
                            _Float16* __restrict__ mW1p) {
    int p = blockIdx.x * blockDim.x + threadIdx.x;      // 131072 threads
    int lane = (p >> 4) & 31, j = p & 15;
    int kin = ((lane & 16) ? 8 : 0) + (j < 8 ? j : j + 8);
    if (p < 32 * 512) {                                 // W1: 64(K pad from 56) x 256
        int t = p >> 9;
        int k = (t >> 4) * 32 + kin;
        int o = (t & 15) * 16 + (lane & 15);
        W1p[p] = (_Float16)(k < 56 ? W1[k * 256 + o] : 0.0f);
    }
    if (p < 128 * 512) {                                // W2: 256 x 256
        int t = p >> 9;
        int k = (t >> 4) * 32 + kin;
        int o = (t & 15) * 16 + (lane & 15);
        W2p[p] = (_Float16)W2[k * 256 + o];
    }
    if (p < 256 * 512) {                                // mem_W1: 8 x 256 x 64
        int t = p >> 9;
        int m = t >> 5, kt = (t >> 2) & 7, ht = t & 3;
        int k = kt * 32 + kin;
        int o = ht * 16 + (lane & 15);
        mW1p[p] = (_Float16)mW1[(m * 256 + k) * 64 + o];
    }
}

// ---- K1: per-node features (32 coord-hash + 16 field-hash + 8 SO2) ---------
__global__ void feat_kernel(const float* __restrict__ R, const float* __restrict__ Z,
                            const float* __restrict__ psi, const float* __restrict__ bpol,
                            const float* __restrict__ ct, const float* __restrict__ ft,
                            const float* __restrict__ so2W, const float* __restrict__ so2b,
                            _Float16* __restrict__ Xf) {
    int n = blockIdx.x * blockDim.x + threadIdx.x;
    if (n >= NN) return;
    float f[64];
    float x = R[n], y = Z[n];
    for (int lev = 0; lev < 16; ++lev) {                // coord grid, T=8192
        float res = (float)RES_TAB[lev];
        float sx = (x * 0.5f + 0.5f) * res, sy = (y * 0.5f + 0.5f) * res;
        float flx = floorf(sx), fly = floorf(sy);
        int ix = (int)flx, iy = (int)fly;
        float fx = sx - flx, fy = sy - fly;
        float a0 = 0.f, a1 = 0.f;
        #pragma unroll
        for (int c = 0; c < 4; ++c) {
            unsigned cx = (unsigned)(ix + (c & 1)), cy = (unsigned)(iy + (c >> 1));
            int idx = (int)(((cx * 2654435761u) ^ (cy * 805459861u)) & (unsigned)(TC - 1));
            float w = ((c & 1) ? fx : 1.f - fx) * ((c >> 1) ? fy : 1.f - fy);
            a0 += w * ct[(lev * TC + idx) * 2 + 0];
            a1 += w * ct[(lev * TC + idx) * 2 + 1];
        }
        f[lev * 2] = a0; f[lev * 2 + 1] = a1;
    }
    float u = psi[n] * 2.0f - 1.0f, v = bpol[n];
    for (int lev = 0; lev < 8; ++lev) {                 // field grid, T=4096
        float res = (float)RES_TAB[lev];
        float sx = (u * 0.5f + 0.5f) * res, sy = (v * 0.5f + 0.5f) * res;
        float flx = floorf(sx), fly = floorf(sy);
        int ix = (int)flx, iy = (int)fly;
        float fx = sx - flx, fy = sy - fly;
        float a0 = 0.f, a1 = 0.f;
        #pragma unroll
        for (int c = 0; c < 4; ++c) {
            unsigned cx = (unsigned)(ix + (c & 1)), cy = (unsigned)(iy + (c >> 1));
            int idx = (int)(((cx * 2654435761u) ^ (cy * 805459861u)) & (unsigned)(TF - 1));
            float w = ((c & 1) ? fx : 1.f - fx) * ((c >> 1) ? fy : 1.f - fy);
            a0 += w * ft[(lev * TF + idx) * 2 + 0];
            a1 += w * ft[(lev * TF + idx) * 2 + 1];
        }
        f[32 + lev * 2] = a0; f[32 + lev * 2 + 1] = a1;
    }
    float th = atan2f(y, x);
    float p[7];
    p[0] = 1.0f;
    p[1] = __cosf(th);        p[2] = __sinf(th);
    p[3] = __cosf(2.f * th);  p[4] = __sinf(2.f * th);
    p[5] = __cosf(3.f * th);  p[6] = __sinf(3.f * th);
    for (int j = 0; j < 8; ++j) {
        float s = so2b[j];
        #pragma unroll
        for (int i = 0; i < 7; ++i) s += p[i] * so2W[i * 8 + j];
        f[48 + j] = s;
    }
    for (int k = 56; k < 64; ++k) f[k] = 0.0f;
    for (int k = 0; k < 64; ++k) Xf[(size_t)n * 64 + k] = (_Float16)f[k];
}

// ---- K2/K3: SIREN GEMM, D = sin(om*(X@W + b)). Block = 16 nodes x 8 waves. -
// A fragments for all K-steps preloaded to VGPRs, reused across out-tiles.
template <int KSTEPS>
__global__ void siren_gemm(const _Float16* __restrict__ X, const _Float16* __restrict__ Wp,
                           const float* __restrict__ bias, const float* __restrict__ om,
                           _Float16* __restrict__ Y) {
    constexpr int KDIM = KSTEPS * 32;
    int wave = threadIdx.x >> 5;     // wave32
    int lane = threadIdx.x & 31;
    int node0 = blockIdx.x << 4;
    int lhi = lane & 16;
    int col = lane & 15;
    float omega = om[0];
    AV a[KSTEPS];
    const _Float16* arow = X + (size_t)(node0 + col) * KDIM + (lhi ? 8 : 0);
    #pragma unroll
    for (int ks = 0; ks < KSTEPS; ++ks) {
        a[ks].h8[0] = *(const v8h*)(arow + ks * 32);
        a[ks].h8[1] = *(const v8h*)(arow + ks * 32 + 16);
    }
    #pragma unroll
    for (int ot = 0; ot < 2; ++ot) {
        int otile = (wave << 1) + ot;
        v8f c = {};
        #pragma unroll
        for (int ks = 0; ks < KSTEPS; ++ks) {
            AV b;                                    // fragment-packed: 2x16B loads
            const _Float16* bp = Wp + ((size_t)(ks * 16 + otile) * 32 + lane) * 16;
            b.h8[0] = *(const v8h*)bp;
            b.h8[1] = *(const v8h*)(bp + 8);
            c = __builtin_amdgcn_wmma_f32_16x16x32_f16(false, a[ks].v, false, b.v,
                                                       short(0), c, false, false);
        }
        #pragma unroll
        for (int r = 0; r < 8; ++r) {
            int node = node0 + r + (lhi ? 8 : 0);
            int out = (otile << 4) + col;
            Y[(size_t)node * 256 + out] = (_Float16)__sinf(omega * (c[r] + bias[out]));
        }
    }
}

// ---- K4: memory heads: 8 waves = 8 heads; 256->64 WMMA, gelu, dot64 via LDS
__global__ void heads_kernel(const _Float16* __restrict__ Tr, const _Float16* __restrict__ mW1p,
                             const float* __restrict__ mb1, const float* __restrict__ mW2,
                             const float* __restrict__ mb2, float* __restrict__ eps) {
    __shared__ float pred[8][16];
    int tid = threadIdx.x;
    int m = tid >> 5;                 // wave = head
    int lane = tid & 31;
    int lhi = lane & 16;
    int col = lane & 15;
    int node0 = blockIdx.x << 4;
    if (tid < 128) pred[tid >> 4][tid & 15] = 0.0f;
    __syncthreads();
    AV a[8];
    const _Float16* arow = Tr + (size_t)(node0 + col) * 256 + (lhi ? 8 : 0);
    #pragma unroll
    for (int ks = 0; ks < 8; ++ks) {
        a[ks].h8[0] = *(const v8h*)(arow + ks * 32);
        a[ks].h8[1] = *(const v8h*)(arow + ks * 32 + 16);
    }
    #pragma unroll
    for (int ht = 0; ht < 4; ++ht) {
        v8f c = {};
        #pragma unroll
        for (int ks = 0; ks < 8; ++ks) {
            AV b;
            const _Float16* bp = mW1p + ((size_t)((m * 8 + ks) * 4 + ht) * 32 + lane) * 16;
            b.h8[0] = *(const v8h*)bp;
            b.h8[1] = *(const v8h*)(bp + 8);
            c = __builtin_amdgcn_wmma_f32_16x16x32_f16(false, a[ks].v, false, b.v,
                                                       short(0), c, false, false);
        }
        #pragma unroll
        for (int r = 0; r < 8; ++r) {
            int nd = r + (lhi ? 8 : 0);
            int hh = (ht << 4) + col;
            float v = c[r] + mb1[m * 64 + hh];
            // gelu(tanh approx): v * sigmoid(1.595769122*(v + 0.044715 v^3))
            float g = v / (1.0f + __expf(-1.5957691216057308f * (v + 0.044715f * v * v * v)));
            atomicAdd(&pred[m][nd], g * mW2[m * 64 + hh]);   // ds_add_f32
        }
    }
    __syncthreads();
    if (tid < 16) {
        float s = 0.0f;
        #pragma unroll
        for (int mm = 0; mm < 8; ++mm) {
            float xx = pred[mm][tid] + mb2[mm];
            s += (xx > 20.0f) ? xx : __logf(1.0f + __expf(xx));
        }
        eps[node0 + tid] = s * 0.125f;
    }
}

// ---- K5: GNN refinement (exact algebraic form) + mask ----------------------
// LayerNorm over the size-1 feature axis: mu == y elementwise, so the
// normalized value is exactly 0 and h = ln_bias[l] after every layer,
// independent of eps_mean / message passing. Numerically live stage -> use
// precise expf/log1pf here.
__global__ void final_kernel(const float* __restrict__ psi, const float* __restrict__ ln_bias,
                             const float* __restrict__ eps_mean, float* __restrict__ out) {
    int n = blockIdx.x * blockDim.x + threadIdx.x;
    if (n >= NN) return;
    (void)eps_mean;   // dead by the exact LN collapse above
    float hb = ln_bias[1];
    float refined = (hb > 20.0f) ? hb : log1pf(expf(hb));
    float t = (1.0f - psi[n]) * 50.0f;
    float mask = 1.0f / (1.0f + expf(-t));
    out[n] = refined * mask;
}

extern "C" void kernel_launch(void* const* d_in, const int* in_sizes, int n_in,
                              void* d_out, int out_size, void* d_ws, size_t ws_size,
                              hipStream_t stream) {
    (void)in_sizes; (void)n_in; (void)out_size; (void)ws_size;
    const float* R    = (const float*)d_in[0];
    const float* Z    = (const float*)d_in[1];
    const float* psi  = (const float*)d_in[2];
    const float* bpol = (const float*)d_in[3];
    const float* ct   = (const float*)d_in[6];
    const float* ft   = (const float*)d_in[7];
    const float* so2W = (const float*)d_in[8];
    const float* so2b = (const float*)d_in[9];
    const float* om1  = (const float*)d_in[10];
    const float* W1   = (const float*)d_in[11];
    const float* b1   = (const float*)d_in[12];
    const float* om2  = (const float*)d_in[13];
    const float* W2   = (const float*)d_in[14];
    const float* b2   = (const float*)d_in[15];
    const float* mW1  = (const float*)d_in[16];
    const float* mb1  = (const float*)d_in[17];
    const float* mW2  = (const float*)d_in[18];
    const float* mb2  = (const float*)d_in[19];
    const float* lnb  = (const float*)d_in[25];
    float* out = (float*)d_out;

    char* wsp = (char*)d_ws;
    size_t off = 0;
    auto carve = [&](size_t bytes) -> char* {
        char* p = wsp + off;
        off += (bytes + 255) & ~(size_t)255;
        return p;
    };
    _Float16* Xf   = (_Float16*)carve((size_t)NN * 64 * 2);     //  8.4 MB
    _Float16* W1p  = (_Float16*)carve(32 * 512 * 2);            //  32 KB packed
    _Float16* W2p  = (_Float16*)carve(128 * 512 * 2);           // 128 KB packed
    _Float16* mW1p = (_Float16*)carve(256 * 512 * 2);           // 256 KB packed
    _Float16* H1   = (_Float16*)carve((size_t)NN * 256 * 2);    // 33.5 MB
    _Float16* Tr   = (_Float16*)carve((size_t)NN * 256 * 2);    // 33.5 MB
    float*    eps  = (float*)carve((size_t)NN * 4);

    pack_kernel<<<512, 256, 0, stream>>>(W1, W2, mW1, W1p, W2p, mW1p);
    feat_kernel<<<NN / 256, 256, 0, stream>>>(R, Z, psi, bpol, ct, ft, so2W, so2b, Xf);
    siren_gemm<2><<<NN / 16, 256, 0, stream>>>(Xf, W1p, b1, om1, H1);
    siren_gemm<8><<<NN / 16, 256, 0, stream>>>(H1, W2p, b2, om2, Tr);
    heads_kernel<<<NN / 16, 256, 0, stream>>>(Tr, mW1p, mb1, mW2, mb2, eps);
    final_kernel<<<NN / 256, 256, 0, stream>>>(psi, lnb, eps, out);
}
